// MultiPathChannel_80736795230988
// MI455X (gfx1250) — matrix operand order repaired and checked by reading, not compiled, not run on hardware
//
#include <hip/hip_runtime.h>

typedef __attribute__((ext_vector_type(16))) _Float16 v16h;
typedef __attribute__((ext_vector_type(8)))  float    v8f;

namespace {
constexpr int kL = 32;
constexpr int kB = 512;
constexpr int kN = 8192;
constexpr int kOutLen = kN + kL - 1;                       // 8223
constexpr int kWaves = 8;                                  // 256 threads, wave32
constexpr int kTilesPerWave = 8;
constexpr int kTilesPerBlock = kWaves * kTilesPerWave;     // 64 tiles
constexpr int kOutPerBlock = kTilesPerBlock * 16;          // 1024 outputs / block
constexpr int kChunks = (kOutLen + kOutPerBlock - 1) / kOutPerBlock;  // 9
constexpr int kXsLen = kOutPerBlock + 48;                  // Hankel halo (max idx 1054)
constexpr int kPairLen = kXsLen / 2;                       // 536 pair entries
}

__device__ __forceinline__ void split_f16(float v, unsigned short& hi, unsigned short& lo) {
  _Float16 h = (_Float16)v;
  _Float16 l = (_Float16)(v - (float)h);
  hi = __builtin_bit_cast(unsigned short, h);
  lo = __builtin_bit_cast(unsigned short, l);
}

// ---------------- Kernel 1: per-batch noise scale ----------------
__global__ __launch_bounds__(256)
void mpc_power_kernel(const float* __restrict__ xr, const float* __restrict__ xi,
                      float* __restrict__ scale) {
  __shared__ float red[256];
  const int b = blockIdx.x;
  float s = 0.f;
  for (int i = threadIdx.x; i < kN; i += 256) {
    float a = xr[(size_t)b * kN + i];
    float c = xi[(size_t)b * kN + i];
    s += a * a + c * c;
  }
  red[threadIdx.x] = s;
  __syncthreads();
  for (int off = 128; off > 0; off >>= 1) {
    if (threadIdx.x < off) red[threadIdx.x] += red[threadIdx.x + off];
    __syncthreads();
  }
  if (threadIdx.x == 0) {
    float x_power = __builtin_sqrtf(red[0]) / __builtin_sqrtf((float)kN);
    // inv_sqrt2 / sqrt(snr) * x_power ; snr = 10^(10/10) = 10
    scale[b] = 0.70710678118654752f / __builtin_sqrtf(10.0f) * x_power;
  }
}

// ---------------- Kernel 2: WMMA complex FIR + noise ----------------
// LDS pair entry layout (64-bit): halfwords [hi(a), hi(b), lo(a), lo(b)] where
//   even copy: (a,b) = (x[2j], x[2j+1]);  odd copy: (a,b) = (x[2j+1], x[2j+2]).
// A lane with Hankel base `base` reads entries base>>1 .. (base>>1)+7 from the
// copy selected by base&1; the low dword of each ds_load_b64 is one f16-hi pair
// in WMMA-B register order, the high dword the matching f16-lo pair.
__global__ __launch_bounds__(256)
void mpc_conv_kernel(const float* __restrict__ xr, const float* __restrict__ xi,
                     const float* __restrict__ hre, const float* __restrict__ him,
                     const float* __restrict__ nre, const float* __restrict__ nim,
                     const float* __restrict__ pdp, const float* __restrict__ scale,
                     float* __restrict__ out) {
  __shared__ unsigned long long xe_r[kPairLen];
  __shared__ unsigned long long xo_r[kPairLen];
  __shared__ unsigned long long xe_i[kPairLen];
  __shared__ unsigned long long xo_i[kPairLen];

  const int b   = blockIdx.y;
  const int T0  = blockIdx.x * kOutPerBlock;
  const int c0  = T0 - (kL - 1);          // out[t] = sum_k x[t+k-31] h[k]
  const int tid = threadIdx.x;

  unsigned short* er = (unsigned short*)xe_r;
  unsigned short* od = (unsigned short*)xo_r;
  unsigned short* ei = (unsigned short*)xe_i;
  unsigned short* oi = (unsigned short*)xo_i;

  for (int i = tid; i < kXsLen; i += 256) {
    int gi = c0 + i;
    float vr = 0.f, vi = 0.f;
    if (gi >= 0 && gi < kN) {
      vr = xr[(size_t)b * kN + gi];
      vi = xi[(size_t)b * kN + gi];
    }
    unsigned short rh, rl, mh, ml;
    split_f16(vr, rh, rl);
    split_f16(vi, mh, ml);
    int je = i >> 1, se = i & 1;
    er[4 * je + se]     = rh;  er[4 * je + 2 + se] = rl;
    ei[4 * je + se]     = mh;  ei[4 * je + 2 + se] = ml;
    if (i >= 1) {
      int jo = (i - 1) >> 1, so = (i & 1) ^ 1;
      od[4 * jo + so]     = rh;  od[4 * jo + 2 + so] = rl;
      oi[4 * jo + so]     = mh;  oi[4 * jo + 2 + so] = ml;
    }
  }

  // Constant filter A-matrices (16x32 f16), built once per block.
  // A_h rows: {hr_hi, hi_hi, hr_lo, hi_lo, 0...}   (multiplies x_real Hankel)
  // A_g rows: {-hi_hi, hr_hi, -hi_lo, hr_lo, 0...} (multiplies x_imag Hankel)
  const int lane = tid & 31;
  const int row  = lane & 15;
  const int sel  = lane >> 4;             // A here: lanes<16 hold taps 0..15
  union V16H { _Float16 h[16]; unsigned u[8]; v16h v; };
  V16H a_h, a_g;
#pragma unroll
  for (int j = 0; j < 16; ++j) {
    int tap = j + 16 * sel;  // -> {0..15} or {16..31}
    float p  = 0.70710678118654752f * pdp[tap];
    float fr = hre[(size_t)b * kL + tap] * p;
    float fi = him[(size_t)b * kL + tap] * p;
    _Float16 frh = (_Float16)fr; _Float16 frl = (_Float16)(fr - (float)frh);
    _Float16 fih = (_Float16)fi; _Float16 fil = (_Float16)(fi - (float)fih);
    _Float16 ah = (_Float16)0.f, ag = (_Float16)0.f;
    if      (row == 0) { ah = frh; ag = -fih; }
    else if (row == 1) { ah = fih; ag =  frh; }
    else if (row == 2) { ah = frl; ag = -fil; }
    else if (row == 3) { ah = fil; ag =  frl; }
    a_h.h[j] = ah;
    a_g.h[j] = ag;
  }
  __syncthreads();

  const float sc = scale[b];
  const int w  = tid >> 5;                // wave id (wave32)
  const int n  = lane & 15;               // B column = output offset in tile
  const int kh = lane >> 4;               // B layout: lanes<16 hold K 0..15
  const unsigned long long* pr = (n & 1) ? xo_r : xe_r;
  const unsigned long long* pq = (n & 1) ? xo_i : xe_i;

  for (int j = 0; j < kTilesPerWave; ++j) {
    const int tile = w * kTilesPerWave + j;
    const int t0   = T0 + tile * 16;
    const int base = tile * 16 + n + kh * 16;   // local Hankel index
    const int j0   = base >> 1;                 // works for both parities

    // Prefetch noise early so HBM latency hides under DS + WMMA.
    const bool act = (lane < 16) && (t0 + lane < kOutLen);
    const size_t no = (size_t)b * kOutLen + (t0 + lane);
    float nrv = 0.f, niv = 0.f;
    if (act) { nrv = nre[no]; niv = nim[no]; }

    V16H brh, brl, bih, bil;
#pragma unroll
    for (int q = 0; q < 8; ++q) {
      unsigned long long e = pr[j0 + q];
      brh.u[q] = (unsigned)e;
      brl.u[q] = (unsigned)(e >> 32);
      unsigned long long f = pq[j0 + q];
      bih.u[q] = (unsigned)f;
      bil.u[q] = (unsigned)(f >> 32);
    }

    // C accumulates (h_hi+h_lo)*(x_hi+x_lo) exactly, in fp32.
    v8f c = {};
    c = __builtin_amdgcn_wmma_f32_16x16x32_f16(false, a_h.v, false, brh.v, (short)0, c, false, false);
    c = __builtin_amdgcn_wmma_f32_16x16x32_f16(false, a_g.v, false, bih.v, (short)0, c, false, false);
    c = __builtin_amdgcn_wmma_f32_16x16x32_f16(false, a_h.v, false, brl.v, (short)0, c, false, false);
    c = __builtin_amdgcn_wmma_f32_16x16x32_f16(false, a_g.v, false, bil.v, (short)0, c, false, false);

    // C rows 0..3 live in VGPRs 0..3 of lanes 0..15 (N = lane).
    if (act) {
      float outr = c[0] + c[2] + nrv * sc;
      float outi = c[1] + c[3] + niv * sc;
      out[no] = outr;                                        // [0, b, o]
      out[((size_t)kB) * (size_t)kOutLen + no] = outi;       // [1, b, o]
    }
  }
}

extern "C" void kernel_launch(void* const* d_in, const int* in_sizes, int n_in,
                              void* d_out, int out_size, void* d_ws, size_t ws_size,
                              hipStream_t stream) {
  (void)in_sizes; (void)n_in; (void)out_size; (void)ws_size;
  const float* xr  = (const float*)d_in[0];
  const float* xi  = (const float*)d_in[1];
  const float* hre = (const float*)d_in[2];
  const float* him = (const float*)d_in[3];
  const float* nre = (const float*)d_in[4];
  const float* nim = (const float*)d_in[5];
  const float* pdp = (const float*)d_in[6];
  float* out   = (float*)d_out;
  float* scale = (float*)d_ws;   // 512 floats of scratch

  mpc_power_kernel<<<kB, 256, 0, stream>>>(xr, xi, scale);
  mpc_conv_kernel<<<dim3(kChunks, kB), 256, 0, stream>>>(
      xr, xi, hre, him, nre, nim, pdp, scale, out);
}
